// SwinTransformerBlock_17377437680450
// MI455X (gfx1250) — compile-verified, matrix operand order checked by reading, not run on hardware
//
#include <hip/hip_runtime.h>
#include <hip/hip_bf16.h>
#include <cstdint>

// ---------------------------------------------------------------------------
// Swin block for MI455X (gfx1250, wave32). Big GEMMs use
// v_wmma_f32_16x16x32_f16 (f16 in, f32 accum). Weight panels are staged per
// workgroup into LDS via the CDNA5 async global->LDS path
// (global_load_async_to_lds_b128 + s_wait_asynccnt) with double buffering
// (panel p+1 in flight while computing panel p), consumed by all 8 waves via
// ds_load_b128. A-fragments stream from row-major f16 global memory and are
// register-pipelined one K-step ahead.
// ---------------------------------------------------------------------------

typedef _Float16 v16h __attribute__((ext_vector_type(16)));
typedef _Float16 h8   __attribute__((ext_vector_type(8)));
typedef float    v8f  __attribute__((ext_vector_type(8)));

__device__ __forceinline__ v8f wmma16(v16h a, v16h b, v8f c) {
    return __builtin_amdgcn_wmma_f32_16x16x32_f16(
        /*neg_a=*/false, a, /*neg_b=*/false, b,
        /*c_mod=*/(short)0, c, /*reuse_a=*/false, /*reuse_b=*/false);
}

// A-fragment (or B-fragment of a transposed-stored matrix) from row-major f16.
__device__ __forceinline__ v16h load_frag_rowmajor(const _Float16* M, int ld,
                                                   int r0, int k0, int lane) {
    int r  = r0 + (lane & 15);
    int kb = k0 + ((lane & 16) ? 8 : 0);
    const _Float16* p = M + (size_t)r * ld + kb;
    union { v16h v; h8 h[2]; } u;
    u.h[0] = *(const h8*)(p);        // K = kb .. kb+7
    u.h[1] = *(const h8*)(p + 16);   // K = kb+16 .. kb+23
    return u.v;
}

// B-fragment from packed layout (global or LDS): 32 lanes x 16 halves.
__device__ __forceinline__ v16h load_frag_packed(const _Float16* Bp, int lane) {
    const _Float16* p = Bp + lane * 16;
    union { v16h v; h8 h[2]; } u;
    u.h[0] = *(const h8*)(p);
    u.h[1] = *(const h8*)(p + 8);
    return u.v;
}

// Async-copy one 16B chunk global -> LDS (ASYNCcnt-tracked).
__device__ __forceinline__ void async_b128(uint32_t lds_addr, const void* gaddr) {
    asm volatile("global_load_async_to_lds_b128 %0, %1, off"
                 :: "v"(lds_addr), "v"(gaddr) : "memory");
}
__device__ __forceinline__ void wait_async0() {
    asm volatile("s_wait_asynccnt 0x0" ::: "memory");
}

// Stage one 12-K-step weight panel (12 x 4KB) into LDS: one b128 per thread
// per K-step, fully coalesced.
__device__ __forceinline__ void stage_panel(const _Float16* __restrict__ Bp,
                                            int NT, int nt0, int kp,
                                            _Float16* dstPanel, int tid) {
#pragma unroll
    for (int k = 0; k < 12; ++k) {
        const _Float16* src = Bp + (((size_t)(kp + k) * NT + nt0) << 9) + tid * 8;
        async_b128((uint32_t)(uintptr_t)(dstPanel + k * 2048 + tid * 8), src);
    }
}

// ---------------------------------------------------------------------------
// Block-cooperative GEMM core. Block = 8 waves; wave w computes the 16x64
// tile at M-rows (mg*8+w)*16, N-cols nq*64.. . npanels x 12 K-steps; LDS
// panels are double-buffered (Bs must hold npanels>1 ? 2 : 1 panels).
// ---------------------------------------------------------------------------
__device__ __forceinline__ void gemm_panel_core(const _Float16* __restrict__ A,
                                                int lda, int m0,
                                                const _Float16* __restrict__ Bp,
                                                int NT, int nt0, int npanels,
                                                _Float16* Bs, int tid, int lane,
                                                v8f acc[4]) {
    stage_panel(Bp, NT, nt0, 0, Bs, tid);
    wait_async0();
    __syncthreads();
    for (int p = 0; p < npanels; ++p) {
        _Float16* cur = Bs + (p & 1) * (12 * 2048);
        if (p + 1 < npanels)   // overlap next panel fetch with compute
            stage_panel(Bp, NT, nt0, (p + 1) * 12,
                        Bs + ((p + 1) & 1) * (12 * 2048), tid);
        v16h a_cur = load_frag_rowmajor(A, lda, m0, p * 384, lane);
#pragma unroll
        for (int ks = 0; ks < 12; ++ks) {
            if (ks < 10)       // prefetch A two K-steps ahead
                __builtin_prefetch(A + (size_t)(m0 + (lane & 15)) * lda +
                                       (p * 12 + ks + 2) * 32, 0, 3);
            v16h a_nxt = a_cur;
            if (ks + 1 < 12)   // register-pipeline the A fragment
                a_nxt = load_frag_rowmajor(A, lda, m0, (p * 12 + ks + 1) * 32, lane);
            // all four B-fragments in distinct registers -> partial dscnt waits
            v16h b0 = load_frag_packed(cur + ks * 2048 +    0, lane);
            v16h b1 = load_frag_packed(cur + ks * 2048 +  512, lane);
            v16h b2 = load_frag_packed(cur + ks * 2048 + 1024, lane);
            v16h b3 = load_frag_packed(cur + ks * 2048 + 1536, lane);
            acc[0] = wmma16(a_cur, b0, acc[0]);
            acc[1] = wmma16(a_cur, b1, acc[1]);
            acc[2] = wmma16(a_cur, b2, acc[2]);
            acc[3] = wmma16(a_cur, b3, acc[3]);
            a_cur = a_nxt;
        }
        if (p + 1 < npanels) wait_async0();
        __syncthreads();       // panel consumed / next panel visible
    }
}

__device__ __forceinline__ float gelu_tanh(float x) {
    float x3 = x * x * x;
    return 0.5f * x * (1.0f + tanhf(0.7978845608028654f * (x + 0.044715f * x3)));
}

// ---------------------------------------------------------------------------
// Weight pre-pack: (K,N) f32 row-major -> f16 fragments [kt][nt][lane][16].
// ---------------------------------------------------------------------------
__global__ __launch_bounds__(256) void pack_b_kernel(const float* __restrict__ W,
                                                     _Float16* __restrict__ P,
                                                     int K, int N) {
    int idx = blockIdx.x * 256 + threadIdx.x;
    if (idx >= K * N) return;
    int e    = idx & 15;
    int lane = (idx >> 4) & 31;
    int frag = idx >> 9;
    int NT   = N >> 4;
    int nt   = frag % NT;
    int kt   = frag / NT;
    int base = (lane & 16) ? 8 : 0;
    int kl   = (e < 8) ? (base + e) : (16 + base + (e - 8));
    int n    = nt * 16 + (lane & 15);
    int k    = kt * 32 + kl;
    P[idx] = (_Float16)W[(size_t)k * N + n];
}

// ---------------------------------------------------------------------------
// LN1 + cyclic shift(-3,-3) + window partition + pad to 64 tokens.
// ---------------------------------------------------------------------------
__global__ __launch_bounds__(128) void ln1_part_kernel(const float* __restrict__ x,
                                                       const float* __restrict__ g,
                                                       const float* __restrict__ b,
                                                       _Float16* __restrict__ xw) {
    int row = blockIdx.x;            // 0 .. 65535  (1024 windows * 64)
    int t = row & 63, w = row >> 6;
    int c = threadIdx.x;
    _Float16* out = xw + (size_t)row * 384;
    if (t >= 49) {                   // padded token rows -> zeros
        out[c] = (_Float16)0.f; out[c + 128] = (_Float16)0.f; out[c + 256] = (_Float16)0.f;
        return;
    }
    int bimg = w >> 6, wim = w & 63;
    int wr = wim >> 3, wc = wim & 7;
    int ty = t / 7, tx = t - ty * 7;
    int Y = wr * 7 + ty + 3; if (Y >= 56) Y -= 56;   // shifted coord -> source
    int X = wc * 7 + tx + 3; if (X >= 56) X -= 56;
    const float* px = x + (((size_t)bimg * 56 + Y) * 56 + X) * 384;
    float v0 = px[c], v1 = px[c + 128], v2 = px[c + 256];
    __shared__ float s1[128], s2[128];
    s1[c] = v0 + v1 + v2;
    s2[c] = v0 * v0 + v1 * v1 + v2 * v2;
    __syncthreads();
    for (int s = 64; s > 0; s >>= 1) {
        if (c < s) { s1[c] += s1[c + s]; s2[c] += s2[c + s]; }
        __syncthreads();
    }
    float mu  = s1[0] * (1.f / 384.f);
    float var = s2[0] * (1.f / 384.f) - mu * mu;
    float rs  = rsqrtf(var + 1e-6f);
    out[c]       = (_Float16)((v0 - mu) * rs * g[c]       + b[c]);
    out[c + 128] = (_Float16)((v1 - mu) * rs * g[c + 128] + b[c + 128]);
    out[c + 256] = (_Float16)((v2 - mu) * rs * g[c + 256] + b[c + 256]);
}

// ---------------------------------------------------------------------------
// LN2: token-major x2 (f32) -> h2 (f16).
// ---------------------------------------------------------------------------
__global__ __launch_bounds__(128) void ln2_kernel(const float* __restrict__ x2,
                                                  const float* __restrict__ g,
                                                  const float* __restrict__ b,
                                                  _Float16* __restrict__ h2) {
    size_t row = blockIdx.x;
    int c = threadIdx.x;
    const float* px = x2 + row * 384;
    float v0 = px[c], v1 = px[c + 128], v2 = px[c + 256];
    __shared__ float s1[128], s2[128];
    s1[c] = v0 + v1 + v2;
    s2[c] = v0 * v0 + v1 * v1 + v2 * v2;
    __syncthreads();
    for (int s = 64; s > 0; s >>= 1) {
        if (c < s) { s1[c] += s1[c + s]; s2[c] += s2[c + s]; }
        __syncthreads();
    }
    float mu  = s1[0] * (1.f / 384.f);
    float var = s2[0] * (1.f / 384.f) - mu * mu;
    float rs  = rsqrtf(var + 1e-6f);
    _Float16* out = h2 + row * 384;
    out[c]       = (_Float16)((v0 - mu) * rs * g[c]       + b[c]);
    out[c + 128] = (_Float16)((v1 - mu) * rs * g[c + 128] + b[c + 128]);
    out[c + 256] = (_Float16)((v2 - mu) * rs * g[c + 256] + b[c + 256]);
}

// ---------------------------------------------------------------------------
// QKV GEMM: (65536 x 384) x (384 x 1152). Epilogue scatters Q (pre-scaled),
// K, and V-transposed.
// ---------------------------------------------------------------------------
__global__ __launch_bounds__(256) void gemm_qkv(const _Float16* __restrict__ A,
                                                const _Float16* __restrict__ Bp,
                                                const float* __restrict__ bias,
                                                _Float16* __restrict__ Qb,
                                                _Float16* __restrict__ Kb,
                                                _Float16* __restrict__ Vt) {
    __shared__ _Float16 Bs[12 * 2048];   // 48 KB, single panel
    int tid = threadIdx.x, wid = tid >> 5, lane = tid & 31;
    int nq = blockIdx.x % 18, mg = blockIdx.x / 18;
    int m0 = (mg * 8 + wid) * 16;
    v8f acc[4] = {{}, {}, {}, {}};
    gemm_panel_core(A, 384, m0, Bp, 72, nq * 4, 1, Bs, tid, lane, acc);
    int rbase = m0 + ((lane & 16) ? 8 : 0);
#pragma unroll
    for (int j = 0; j < 4; ++j) {
        int col = nq * 64 + j * 16 + (lane & 15);
        float bv = bias[col];
#pragma unroll
        for (int r = 0; r < 8; ++r) {
            int row = rbase + r;
            int w = row >> 6, t = row & 63;
            float v = acc[j][r] + bv;
            if (col < 384) {
                int hh = col >> 5, d = col & 31;
                Qb[(((size_t)w * 12 + hh) * 64 + t) * 32 + d] =
                    (_Float16)(v * 0.17677669529663687f);
            } else if (col < 768) {
                int c2 = col - 384, hh = c2 >> 5, d = c2 & 31;
                Kb[(((size_t)w * 12 + hh) * 64 + t) * 32 + d] = (_Float16)v;
            } else {
                int c2 = col - 768, hh = c2 >> 5, d = c2 & 31;
                Vt[(((size_t)w * 12 + hh) * 32 + d) * 64 + t] = (_Float16)v;
            }
        }
    }
}

// ---------------------------------------------------------------------------
// Attention per (window, head): S = Q K^T (WMMA), softmax with rel-pos bias +
// shift mask + pad mask, O = P V (WMMA, A from LDS).
// ---------------------------------------------------------------------------
__global__ __launch_bounds__(256) void attn_kernel(const _Float16* __restrict__ Qb,
                                                   const _Float16* __restrict__ Kb,
                                                   const _Float16* __restrict__ Vt,
                                                   const float* __restrict__ relT,
                                                   _Float16* __restrict__ outw) {
    int wh = blockIdx.x;
    int w  = wh / 12, hh = wh % 12;
    int tid = threadIdx.x, wid = tid >> 5, lane = tid & 31;
    __shared__ float    S[64][68];
    __shared__ _Float16 P[64][64];
    const _Float16* Qbase = Qb + (size_t)wh * 2048;
    const _Float16* Kbase = Kb + (size_t)wh * 2048;
    const _Float16* Vbase = Vt + (size_t)wh * 2048;
    int rb = (lane & 16) ? 8 : 0;

    // S = Q * K^T : 16 tiles of 16x16, 8 waves x 2 tiles, single K-step (K=32).
#pragma unroll
    for (int tl = 0; tl < 2; ++tl) {
        int tau = wid * 2 + tl;
        int mt = tau >> 2, nt = tau & 3;
        v16h a = load_frag_rowmajor(Qbase, 32, mt * 16, 0, lane);
        v16h b = load_frag_rowmajor(Kbase, 32, nt * 16, 0, lane);  // B = K^T
        v8f  c = {};
        c = wmma16(a, b, c);
#pragma unroll
        for (int r = 0; r < 8; ++r)
            S[mt * 16 + rb + r][nt * 16 + (lane & 15)] = c[r];
    }
    __syncthreads();

    // Row softmax with rel-pos bias, shifted-window mask, pad mask.
    if (tid < 64) {
        int i = tid;
        if (i >= 49) {
            for (int j = 0; j < 64; ++j) P[i][j] = (_Float16)0.f;
        } else {
            int yi = i / 7, xi = i - yi * 7;
            int wim = w & 63, wr = wim >> 3, wc = wim & 7;
            int ys = wr * 7 + yi, xs = wc * 7 + xi;
            int regi = (ys < 49 ? 0 : (ys < 53 ? 1 : 2)) * 3 +
                       (xs < 49 ? 0 : (xs < 53 ? 1 : 2));
            float mx = -1e30f;
            for (int j = 0; j < 64; ++j) {
                float v;
                if (j < 49) {
                    int yj = j / 7, xj = j - yj * 7;
                    int ysj = wr * 7 + yj, xsj = wc * 7 + xj;
                    int regj = (ysj < 49 ? 0 : (ysj < 53 ? 1 : 2)) * 3 +
                               (xsj < 49 ? 0 : (xsj < 53 ? 1 : 2));
                    int ridx = (yi - yj + 6) * 13 + (xi - xj + 6);
                    v = S[i][j] + relT[ridx * 12 + hh] +
                        (regi != regj ? -100.f : 0.f);
                } else {
                    v = -1e30f;                 // padded keys
                }
                S[i][j] = v;
                mx = fmaxf(mx, v);
            }
            float sum = 0.f;
            for (int j = 0; j < 64; ++j) {
                float e = __expf(S[i][j] - mx);
                S[i][j] = e;
                sum += e;
            }
            float inv = 1.f / sum;
            for (int j = 0; j < 64; ++j) P[i][j] = (_Float16)(S[i][j] * inv);
        }
    }
    __syncthreads();

    // O = P * V : 8 tiles (4 M x 2 N), 2 K-steps (K=64). A from LDS, B from Vt.
    {
        int mt = wid >> 1, nt = wid & 1;
        v8f c = {};
#pragma unroll
        for (int ks = 0; ks < 2; ++ks) {
            v16h a = load_frag_rowmajor(&P[0][0], 64, mt * 16, ks * 32, lane);
            v16h b = load_frag_rowmajor(Vbase, 64, nt * 16, ks * 32, lane);
            c = wmma16(a, b, c);
        }
#pragma unroll
        for (int r = 0; r < 8; ++r) {
            int row = mt * 16 + rb + r;
            outw[((size_t)w * 64 + row) * 384 + hh * 32 + nt * 16 + (lane & 15)] =
                (_Float16)c[r];
        }
    }
}

// ---------------------------------------------------------------------------
// proj GEMM (65536 x 384) x (384 x 384) -> window-reverse + unshift + residual.
// ---------------------------------------------------------------------------
__global__ __launch_bounds__(256) void gemm_proj(const _Float16* __restrict__ A,
                                                 const _Float16* __restrict__ Bp,
                                                 const float* __restrict__ bias,
                                                 const float* __restrict__ xin,
                                                 float* __restrict__ x2) {
    __shared__ _Float16 Bs[12 * 2048];
    int tid = threadIdx.x, wid = tid >> 5, lane = tid & 31;
    int nq = blockIdx.x % 6, mg = blockIdx.x / 6;
    int m0 = (mg * 8 + wid) * 16;
    v8f acc[4] = {{}, {}, {}, {}};
    gemm_panel_core(A, 384, m0, Bp, 24, nq * 4, 1, Bs, tid, lane, acc);
    int rbase = m0 + ((lane & 16) ? 8 : 0);
#pragma unroll
    for (int j = 0; j < 4; ++j) {
        int col = nq * 64 + j * 16 + (lane & 15);
        float bv = bias[col];
#pragma unroll
        for (int r = 0; r < 8; ++r) {
            int row = rbase + r;
            int w = row >> 6, t = row & 63;
            if (t < 49) {
                int bimg = w >> 6, wim = w & 63;
                int wr = wim >> 3, wc = wim & 7;
                int ty = t / 7, tx = t - ty * 7;
                int Y = wr * 7 + ty + 3; if (Y >= 56) Y -= 56;
                int X = wc * 7 + tx + 3; if (X >= 56) X -= 56;
                size_t idx = (((size_t)bimg * 56 + Y) * 56 + X) * 384 + col;
                x2[idx] = xin[idx] + acc[j][r] + bv;
            }
        }
    }
}

// ---------------------------------------------------------------------------
// fc1 GEMM (50176 x 384) x (384 x 1536), GELU, f16 out.
// ---------------------------------------------------------------------------
__global__ __launch_bounds__(256) void gemm_fc1(const _Float16* __restrict__ A,
                                                const _Float16* __restrict__ Bp,
                                                const float* __restrict__ bias,
                                                _Float16* __restrict__ g) {
    __shared__ _Float16 Bs[12 * 2048];
    int tid = threadIdx.x, wid = tid >> 5, lane = tid & 31;
    int nq = blockIdx.x % 24, mg = blockIdx.x / 24;
    int m0 = (mg * 8 + wid) * 16;
    v8f acc[4] = {{}, {}, {}, {}};
    gemm_panel_core(A, 384, m0, Bp, 96, nq * 4, 1, Bs, tid, lane, acc);
    int rbase = m0 + ((lane & 16) ? 8 : 0);
#pragma unroll
    for (int j = 0; j < 4; ++j) {
        int col = nq * 64 + j * 16 + (lane & 15);
        float bv = bias[col];
#pragma unroll
        for (int r = 0; r < 8; ++r) {
            int row = rbase + r;
            g[(size_t)row * 1536 + col] = (_Float16)gelu_tanh(acc[j][r] + bv);
        }
    }
}

// ---------------------------------------------------------------------------
// fc2 GEMM (50176 x 1536) x (1536 x 384), + residual x2 -> d_out (f32).
// K = 1536 -> 4 panels, double-buffered in LDS (96 KB).
// ---------------------------------------------------------------------------
__global__ __launch_bounds__(256) void gemm_fc2(const _Float16* __restrict__ A,
                                                const _Float16* __restrict__ Bp,
                                                const float* __restrict__ bias,
                                                const float* __restrict__ x2,
                                                float* __restrict__ out) {
    __shared__ _Float16 Bs[2 * 12 * 2048];   // 96 KB, double-buffered
    int tid = threadIdx.x, wid = tid >> 5, lane = tid & 31;
    int nq = blockIdx.x % 6, mg = blockIdx.x / 6;
    int m0 = (mg * 8 + wid) * 16;
    v8f acc[4] = {{}, {}, {}, {}};
    gemm_panel_core(A, 1536, m0, Bp, 24, nq * 4, 4, Bs, tid, lane, acc);
    int rbase = m0 + ((lane & 16) ? 8 : 0);
#pragma unroll
    for (int j = 0; j < 4; ++j) {
        int col = nq * 64 + j * 16 + (lane & 15);
        float bv = bias[col];
#pragma unroll
        for (int r = 0; r < 8; ++r) {
            int row = rbase + r;
            size_t idx = (size_t)row * 384 + col;
            out[idx] = x2[idx] + acc[j][r] + bv;
        }
    }
}

// ---------------------------------------------------------------------------
extern "C" void kernel_launch(void* const* d_in, const int* in_sizes, int n_in,
                              void* d_out, int out_size, void* d_ws, size_t ws_size,
                              hipStream_t stream) {
    (void)in_sizes; (void)n_in; (void)out_size; (void)ws_size;
    const float* x     = (const float*)d_in[0];
    const float* n1s   = (const float*)d_in[1];
    const float* n1b   = (const float*)d_in[2];
    const float* qkvW  = (const float*)d_in[3];
    const float* qkvB  = (const float*)d_in[4];
    const float* relT  = (const float*)d_in[5];
    const float* projW = (const float*)d_in[6];
    const float* projB = (const float*)d_in[7];
    const float* n2s   = (const float*)d_in[8];
    const float* n2b   = (const float*)d_in[9];
    const float* fc1W  = (const float*)d_in[10];
    const float* fc1B  = (const float*)d_in[11];
    const float* fc2W  = (const float*)d_in[12];
    const float* fc2B  = (const float*)d_in[13];
    float* out = (float*)d_out;

    char* ws = (char*)d_ws;
    size_t off = 0;
    auto take = [&](size_t bytes) -> void* {
        void* p = ws + off;
        off = (off + bytes + 255) & ~(size_t)255;
        return p;
    };
    _Float16* xw    = (_Float16*)take(65536ull * 384 * 2);   // LN1+shift+partition, padded
    _Float16* Qb    = (_Float16*)take(12288ull * 2048 * 2);  // (w,h,64,32)
    _Float16* Kb    = (_Float16*)take(12288ull * 2048 * 2);  // (w,h,64,32)
    _Float16* Vt    = (_Float16*)take(12288ull * 2048 * 2);  // (w,h,32,64) transposed
    _Float16* outw  = (_Float16*)take(65536ull * 384 * 2);   // attention output (window rows)
    float*    x2    = (float*)   take(19267584ull * 4);      // post-attn residual (B,H,W,C)
    _Float16* h2    = (_Float16*)take(19267584ull * 2);      // LN2 out
    _Float16* gbuf  = (_Float16*)take(77070336ull * 2);      // GELU(fc1) out
    _Float16* pQKV  = (_Float16*)take(442368ull * 2);
    _Float16* pPROJ = (_Float16*)take(147456ull * 2);
    _Float16* pFC1  = (_Float16*)take(589824ull * 2);
    _Float16* pFC2  = (_Float16*)take(589824ull * 2);

    // 1) pack weights into WMMA B-fragment layout
    pack_b_kernel<<<(442368 + 255) / 256, 256, 0, stream>>>(qkvW, pQKV, 384, 1152);
    pack_b_kernel<<<(147456 + 255) / 256, 256, 0, stream>>>(projW, pPROJ, 384, 384);
    pack_b_kernel<<<(589824 + 255) / 256, 256, 0, stream>>>(fc1W, pFC1, 384, 1536);
    pack_b_kernel<<<(589824 + 255) / 256, 256, 0, stream>>>(fc2W, pFC2, 1536, 384);

    // 2) LN1 + shift + window partition (pad 49 -> 64 tokens with zeros)
    ln1_part_kernel<<<65536, 128, 0, stream>>>(x, n1s, n1b, xw);

    // 3) QKV projection (WMMA, async-LDS weight panel)
    gemm_qkv<<<9216, 256, 0, stream>>>(xw, pQKV, qkvB, Qb, Kb, Vt);

    // 4) windowed attention (WMMA QK^T and PV, softmax in LDS)
    attn_kernel<<<12288, 256, 0, stream>>>(Qb, Kb, Vt, relT, outw);

    // 5) output projection + window reverse + unshift + residual
    gemm_proj<<<3072, 256, 0, stream>>>(outw, pPROJ, projB, x, x2);

    // 6) LN2
    ln2_kernel<<<50176, 128, 0, stream>>>(x2, n2s, n2b, h2);

    // 7) fc1 + GELU
    gemm_fc1<<<9408, 256, 0, stream>>>(h2, pFC1, fc1B, gbuf);

    // 8) fc2 + residual (double-buffered async panels) -> d_out
    gemm_fc2<<<2352, 256, 0, stream>>>(gbuf, pFC2, fc2B, x2, out);
}
